// HMM_EM_64776696758976
// MI455X (gfx1250) — matrix operand order repaired and checked by reading, not compiled, not run on hardware
//
#include <hip/hip_runtime.h>
#include <hip/hip_bf16.h>
#include <math.h>

typedef __attribute__((ext_vector_type(16))) _Float16 v16h;
typedef __attribute__((ext_vector_type(8)))  _Float16 v8h;
typedef __attribute__((ext_vector_type(8)))  float    v8f;

#define SEQ_LEN 1024
#define BATCH   256
#define ZDIM    128
#define XDIM    10000

// ---------------- workspace layout (bytes) ----------------
// 0      : lseE   f32[128]
// 512    : logpi  f32[128]
// 1024   : Tt     f16[128*128]   (Tt[i*128+j] = exp(logT[j,i]), i.e. B matrix K x N)
// 33792  : m_part f32[40*128]
// 54272  : s_part f32[40*128]

// Kernel A: partial column logsumexp of emit [XDIM x ZDIM], 40 blocks x 250 rows.
__global__ void hmm_lse_partial(const float* __restrict__ emit,
                                float* __restrict__ m_part,
                                float* __restrict__ s_part) {
    int z = threadIdx.x;          // 128 threads
    int g = blockIdx.x;           // 40 blocks
    int x0 = g * 250;
    float m = -INFINITY, s = 0.f;
    for (int x = x0; x < x0 + 250; ++x) {
        float v = emit[(size_t)x * ZDIM + z];
        float nm = fmaxf(m, v);
        s = s * __expf(m - nm) + __expf(v - nm);
        m = nm;
    }
    m_part[g * ZDIM + z] = m;
    s_part[g * ZDIM + z] = s;
}

// Kernel B: combine partials -> lseE; logpi; Tt = exp(log_softmax(T, axis=0))^T as f16.
__global__ void hmm_precomp(const float* __restrict__ T,
                            const float* __restrict__ pi,
                            const float* __restrict__ m_part,
                            const float* __restrict__ s_part,
                            float* __restrict__ lseE,
                            float* __restrict__ logpi,
                            _Float16* __restrict__ Tt) {
    __shared__ float sh[128];
    __shared__ float lseT_sh[128];
    int z = threadIdx.x;          // 128 threads

    // combine emission partials
    float m = -INFINITY, s = 0.f;
    for (int g = 0; g < 40; ++g) {
        float mp = m_part[g * ZDIM + z];
        float sp = s_part[g * ZDIM + z];
        float nm = fmaxf(m, mp);
        s = s * __expf(m - nm) + sp * __expf(mp - nm);
        m = nm;
    }
    lseE[z] = m + __logf(s);

    // logpi = pi - logsumexp(pi)
    float pv = pi[z];
    sh[z] = pv; __syncthreads();
    for (int off = 64; off > 0; off >>= 1) {
        if (z < off) sh[z] = fmaxf(sh[z], sh[z + off]);
        __syncthreads();
    }
    float pmax = sh[0]; __syncthreads();
    sh[z] = __expf(pv - pmax); __syncthreads();
    for (int off = 64; off > 0; off >>= 1) {
        if (z < off) sh[z] += sh[z + off];
        __syncthreads();
    }
    float plse = pmax + __logf(sh[0]); __syncthreads();
    logpi[z] = pv - plse;

    // lseT[i] = logsumexp_j T[j,i]  (softmax over axis 0)
    float mt = -INFINITY, st = 0.f;
    for (int j = 0; j < ZDIM; ++j) {
        float v = T[j * ZDIM + z];
        float nm = fmaxf(mt, v);
        st = st * __expf(mt - nm) + __expf(v - nm);
        mt = nm;
    }
    lseT_sh[z] = mt + __logf(st);
    __syncthreads();

    // Tt[i*128 + j] = exp(T[j,i] - lseT[i])   (B matrix, K=i rows, N=j cols)
    for (int idx = z; idx < ZDIM * ZDIM; idx += 128) {
        int i = idx >> 7, j = idx & 127;
        Tt[idx] = (_Float16)__expf(T[j * ZDIM + i] - lseT_sh[i]);
    }
}

// Main persistent kernel: 16 blocks x 256 threads (8 wave32s), 16 batches per block.
__global__ __launch_bounds__(256)
void hmm_forward(const int* __restrict__ tokens,
                 const float* __restrict__ emit,
                 const float* __restrict__ lseE,
                 const float* __restrict__ logpi,
                 const _Float16* __restrict__ Tt,
                 float* __restrict__ out) {
    __shared__ __align__(16) float    alpha_lds[16 * 132];  // padded rows (132 f32)
    __shared__ __align__(16) _Float16 p_lds[16 * 136];      // padded rows (136 f16 = 272B)
    __shared__ float m_lds[16];

    const int tid  = threadIdx.x;
    const int w    = tid >> 5;            // wave 0..7 -> owns j-tile [16w,16w+16)
    const int lane = tid & 31;
    const int bbase = blockIdx.x << 4;    // 16 batches per block
    const int Ncol = (w << 4) + (lane & 15);   // global j this lane's C column maps to
    const int kb2  = (lane < 16) ? 0 : 16;     // (unused; B fragment uses offA table order)
    const int offA = (lane < 16) ? 0 : 8;      // A fragment K offset per half-wave
    const int M    = lane & 15;                // A fragment row
    const int rb   = tid >> 4;                 // 0..15 : batch row for reductions
    const int seg  = tid & 15;                 // 8-wide segment within row

    const float lseE_j = lseE[Ncol];
    const float lpi_j  = logpi[Ncol];

    // ---- B fragments: constant across all time steps; load once ----
    v16h bfrag[4];
#pragma unroll
    for (int kk = 0; kk < 4; ++kk) {
        v16h b;
#pragma unroll
        for (int e = 0; e < 16; ++e)
            b[e] = Tt[(kk * 32 + offA + e + (e >= 8 ? 8 : 0)) * ZDIM + Ncol];
        bfrag[kk] = b;
    }

    // ---- t = 0 : alpha0 = obs0 + logpi ----
#pragma unroll
    for (int r = 0; r < 8; ++r) {
        int brow = (lane < 16) ? r : (r + 8);
        int tok  = tokens[bbase + brow];
        alpha_lds[brow * 132 + Ncol] =
            (emit[(size_t)tok * ZDIM + Ncol] - lseE_j) + lpi_j;
    }
    // prefetch obs for t = 1
    float obs_cur[8];
#pragma unroll
    for (int r = 0; r < 8; ++r) {
        int brow = (lane < 16) ? r : (r + 8);
        int tok  = tokens[BATCH + bbase + brow];
        obs_cur[r] = emit[(size_t)tok * ZDIM + Ncol] - lseE_j;
    }
    __syncthreads();

    for (int t = 1; t < SEQ_LEN; ++t) {
        // -- phase 1: row max m_b, p = exp(alpha - m) as f16 into A-staging LDS --
        const float* arow = &alpha_lds[rb * 132 + seg * 8];
        float av[8];
#pragma unroll
        for (int k = 0; k < 8; ++k) av[k] = arow[k];
        float mx = av[0];
#pragma unroll
        for (int k = 1; k < 8; ++k) mx = fmaxf(mx, av[k]);
#pragma unroll
        for (int d = 1; d < 16; d <<= 1) mx = fmaxf(mx, __shfl_xor(mx, d, 32));
        if (seg == 0) m_lds[rb] = mx;
        v8h ph;
#pragma unroll
        for (int k = 0; k < 8; ++k) ph[k] = (_Float16)__expf(av[k] - mx);
        *(v8h*)(&p_lds[rb * 136 + seg * 8]) = ph;
        __syncthreads();

        // -- prefetch obs for t+1 (latency hides across WMMA/epilogue/barrier) --
        int tn = (t < SEQ_LEN - 1) ? (t + 1) : t;
        float obs_nx[8];
#pragma unroll
        for (int r = 0; r < 8; ++r) {
            int brow = (lane < 16) ? r : (r + 8);
            int tok  = tokens[tn * BATCH + bbase + brow];
            obs_nx[r] = emit[(size_t)tok * ZDIM + Ncol] - lseE_j;
        }

        // -- WMMA: s_tile(16x16) = P(16x128) x Tt(128 x [16w..16w+16)) --
        const _Float16* prow = &p_lds[M * 136];
        v8f c = {0.f, 0.f, 0.f, 0.f, 0.f, 0.f, 0.f, 0.f};
#pragma unroll
        for (int kk = 0; kk < 4; ++kk) {
            v8h lo = *(const v8h*)(prow + kk * 32 + offA);
            v8h hi = *(const v8h*)(prow + kk * 32 + 16 + offA);
            v16h a;
#pragma unroll
            for (int e = 0; e < 8; ++e) { a[e] = lo[e]; a[e + 8] = hi[e]; }
            c = __builtin_amdgcn_wmma_f32_16x16x32_f16(
                    false, a, false, bfrag[kk], (short)0, c, false, false);
        }

        // -- epilogue: alpha_new = obs + log(s) + m --
#pragma unroll
        for (int r = 0; r < 8; ++r) {
            int brow = (lane < 16) ? r : (r + 8);
            alpha_lds[brow * 132 + Ncol] = obs_cur[r] + __logf(c[r]) + m_lds[brow];
        }
        __syncthreads();
#pragma unroll
        for (int r = 0; r < 8; ++r) obs_cur[r] = obs_nx[r];
    }

    // ---- final: out[b] = -(m + log(sum_j exp(alpha - m))) ----
    const float* frow = &alpha_lds[rb * 132 + seg * 8];
    float fv[8];
#pragma unroll
    for (int k = 0; k < 8; ++k) fv[k] = frow[k];
    float fm = fv[0];
#pragma unroll
    for (int k = 1; k < 8; ++k) fm = fmaxf(fm, fv[k]);
#pragma unroll
    for (int d = 1; d < 16; d <<= 1) fm = fmaxf(fm, __shfl_xor(fm, d, 32));
    float fs = 0.f;
#pragma unroll
    for (int k = 0; k < 8; ++k) fs += __expf(fv[k] - fm);
#pragma unroll
    for (int d = 1; d < 16; d <<= 1) fs += __shfl_xor(fs, d, 32);
    if (seg == 0) out[bbase + rb] = -(fm + __logf(fs));
}

extern "C" void kernel_launch(void* const* d_in, const int* in_sizes, int n_in,
                              void* d_out, int out_size, void* d_ws, size_t ws_size,
                              hipStream_t stream) {
    const int*   tokens = (const int*)d_in[0];     // [1024,256] int32
    const float* T      = (const float*)d_in[1];   // [128,128]
    const float* pi     = (const float*)d_in[2];   // [128]
    const float* emit   = (const float*)d_in[3];   // [10000,128]
    float* out = (float*)d_out;                    // [256]

    char* ws = (char*)d_ws;
    float*    lseE   = (float*)(ws + 0);
    float*    logpi  = (float*)(ws + 512);
    _Float16* Tt     = (_Float16*)(ws + 1024);
    float*    m_part = (float*)(ws + 33792);
    float*    s_part = (float*)(ws + 54272);

    hmm_lse_partial<<<40, 128, 0, stream>>>(emit, m_part, s_part);
    hmm_precomp<<<1, 128, 0, stream>>>(T, pi, m_part, s_part, lseE, logpi, Tt);
    hmm_forward<<<16, 256, 0, stream>>>(tokens, emit, lseE, logpi, Tt, out);
}
